// ResGCNBlock_66812511257312
// MI455X (gfx1250) — matrix operand order repaired and checked by reading, not compiled, not run on hardware
//
#include <hip/hip_runtime.h>
#include <hip/hip_bf16.h>
#include <stdint.h>

#define DIM 128
#define LDS_STRIDE 136              // DIM + 8 bf16 pad -> conflict-free b128 LDS reads
#define ROWS_PER_BLOCK 64           // 4 row tiles of 16

typedef __attribute__((ext_vector_type(16))) __bf16 v16bf;
typedef __attribute__((ext_vector_type(8)))  float  v8f;

// ---------------------------------------------------------------------------
// Kernel 1a: init degree with self-loop contribution (deg = 1)
// ---------------------------------------------------------------------------
__global__ void k_deg_init(float* __restrict__ deg, int n) {
    int i = blockIdx.x * blockDim.x + threadIdx.x;
    if (i < n) deg[i] = 1.0f;
}

// ---------------------------------------------------------------------------
// Kernel 1b: deg[dst] += 1 for every edge
// ---------------------------------------------------------------------------
__global__ void k_deg_count(const long long* __restrict__ ei, float* __restrict__ deg,
                            int nedges) {
    int e = blockIdx.x * blockDim.x + threadIdx.x;
    if (e < nedges) {
        long long d = ei[(long long)nedges + e];  // edge_index[1][e]
        atomicAdd(&deg[d], 1.0f);
    }
}

// ---------------------------------------------------------------------------
// Kernel 2: h = x @ W via v_wmma_f32_16x16x32_bf16.
// One block = 64 rows (4 row tiles). 8 waves, one per 16-column tile of W.
// W is staged TRANSPOSED ([n][k]) so every lane's B fragment is contiguous
// (2x ds_load_b128). Each wave keeps 4 accumulators and reuses the B
// fragment across 4 WMMAs per k-step -> 16 WMMAs per wave.
// ---------------------------------------------------------------------------
__global__ __launch_bounds__(256) void k_gemm_wmma(const float* __restrict__ x,
                                                   const float* __restrict__ W,
                                                   float* __restrict__ h, int N) {
    __shared__ __bf16 sWt[DIM * LDS_STRIDE];             // [n][k] transposed, ~34 KB
    __shared__ __bf16 sX[ROWS_PER_BLOCK * LDS_STRIDE];   // [m][k], ~17 KB

    const int tid     = threadIdx.x;
    const int rowBase = blockIdx.x * ROWS_PER_BLOCK;

    // Stage W transposed as bf16 (64 elems / thread). W is [k][n] row-major.
    for (int i = tid; i < DIM * DIM; i += 256) {
        int k = i >> 7, n = i & (DIM - 1);
        sWt[n * LDS_STRIDE + k] = (__bf16)W[i];
    }
    // Stage the 64-row x tile as bf16, zero-filling rows past N.
    for (int i = tid; i < ROWS_PER_BLOCK * DIM; i += 256) {
        int m = i >> 7, k = i & (DIM - 1);
        int row = rowBase + m;
        sX[m * LDS_STRIDE + k] =
            (row < N) ? (__bf16)x[(size_t)row * DIM + k] : (__bf16)0.0f;
    }
    __syncthreads();

    const int wave = tid >> 5;       // 0..7 -> output column tile
    const int lane = tid & 31;
    const int r    = lane & 15;      // row (A) / col (B) within tile
    const int half = lane >> 4;      // lane half selects K interleave
    const int nCol = wave * 16 + r;

    v8f acc0 = {}, acc1 = {}, acc2 = {}, acc3 = {};

    #pragma unroll
    for (int kk = 0; kk < DIM; kk += 32) {
        // B fragment (32x16 bf16): half 0 lanes hold K {kk+0..7, kk+16..23},
        // half 1 lanes hold K {kk+8..15, kk+24..31} (matches A).
        v16bf b;
        const int kB = kk + half * 8;
        #pragma unroll
        for (int j = 0; j < 16; ++j) b[j] = sWt[nCol * LDS_STRIDE + kB + j + (j & 8)];

        // A fragments for 4 row tiles: half 0 holds K {kk+0..7, kk+16..23},
        // half 1 holds K {kk+8..15, kk+24..31}.
        const int kA = kk + half * 8;
        v16bf a0, a1, a2, a3;
        #pragma unroll
        for (int j = 0; j < 8; ++j) {
            a0[j]     = sX[(r +  0) * LDS_STRIDE + kA + j];
            a0[j + 8] = sX[(r +  0) * LDS_STRIDE + kA + 16 + j];
            a1[j]     = sX[(r + 16) * LDS_STRIDE + kA + j];
            a1[j + 8] = sX[(r + 16) * LDS_STRIDE + kA + 16 + j];
            a2[j]     = sX[(r + 32) * LDS_STRIDE + kA + j];
            a2[j + 8] = sX[(r + 32) * LDS_STRIDE + kA + 16 + j];
            a3[j]     = sX[(r + 48) * LDS_STRIDE + kA + j];
            a3[j + 8] = sX[(r + 48) * LDS_STRIDE + kA + 16 + j];
        }

        acc0 = __builtin_amdgcn_wmma_f32_16x16x32_bf16(false, a0, false, b, (short)0, acc0, false, false);
        acc1 = __builtin_amdgcn_wmma_f32_16x16x32_bf16(false, a1, false, b, (short)0, acc1, false, false);
        acc2 = __builtin_amdgcn_wmma_f32_16x16x32_bf16(false, a2, false, b, (short)0, acc2, false, false);
        acc3 = __builtin_amdgcn_wmma_f32_16x16x32_bf16(false, a3, false, b, (short)0, acc3, false, false);
    }

    // C layout: VGPR i -> M = i + half*8 within each 16-row subtile, N = nCol.
    #pragma unroll
    for (int i = 0; i < 8; ++i) {
        int m0 = rowBase +  0 + i + half * 8;
        int m1 = rowBase + 16 + i + half * 8;
        int m2 = rowBase + 32 + i + half * 8;
        int m3 = rowBase + 48 + i + half * 8;
        if (m0 < N) h[(size_t)m0 * DIM + nCol] = acc0[i];
        if (m1 < N) h[(size_t)m1 * DIM + nCol] = acc1[i];
        if (m2 < N) h[(size_t)m2 * DIM + nCol] = acc2[i];
        if (m3 < N) h[(size_t)m3 * DIM + nCol] = acc3[i];
    }
}

// ---------------------------------------------------------------------------
// Kernel 3: seed accumulator with self-loop term: accum = h * dis[i]^2 = h/deg
// ---------------------------------------------------------------------------
__global__ void k_selfloop(const float* __restrict__ h, const float* __restrict__ deg,
                           float* __restrict__ accum, int total) {
    int idx = blockIdx.x * blockDim.x + threadIdx.x;
    if (idx < total) {
        int node = idx >> 7;  // / DIM
        accum[idx] = h[idx] / deg[node];
    }
}

// ---------------------------------------------------------------------------
// Kernel 4: edge scatter. One wave per edge; each lane moves 4 floats.
// accum[dst] += h[src] * rsqrt(deg[src]*deg[dst])
// ---------------------------------------------------------------------------
__global__ __launch_bounds__(256) void k_scatter(const long long* __restrict__ ei,
                                                 const float* __restrict__ h,
                                                 const float* __restrict__ deg,
                                                 float* __restrict__ accum,
                                                 int nedges) {
    int gid  = blockIdx.x * blockDim.x + threadIdx.x;
    int e    = gid >> 5;
    int lane = gid & 31;
    if (e >= nedges) return;

    long long s = ei[e];                       // edge_index[0][e]
    long long d = ei[(long long)nedges + e];   // edge_index[1][e]
    float norm  = __frsqrt_rn(deg[s] * deg[d]);

    const float4* hp = (const float4*)(h + (size_t)s * DIM);
    float4 v = hp[lane];                       // 32 lanes * 4 = 128 elems
    float* ap = accum + (size_t)d * DIM + lane * 4;
    atomicAdd(ap + 0, v.x * norm);
    atomicAdd(ap + 1, v.y * norm);
    atomicAdd(ap + 2, v.z * norm);
    atomicAdd(ap + 3, v.w * norm);
}

// ---------------------------------------------------------------------------
// Kernel 5: out = relu(accum + b) * dropout / (1-p) + x
// Deterministic hash-based Bernoulli(0.5) (compile-only loop; JAX threefry not
// bit-reproducible here, but the mask is fixed across calls).
// ---------------------------------------------------------------------------
__device__ __forceinline__ uint32_t hash_u32(uint32_t v) {
    v ^= 0x9E3779B9u;
    v *= 0x85EBCA6Bu; v ^= v >> 13;
    v *= 0xC2B2AE35u; v ^= v >> 16;
    return v;
}

__global__ void k_finalize(const float* __restrict__ accum, const float* __restrict__ b,
                           const float* __restrict__ x, float* __restrict__ out,
                           int total) {
    int idx = blockIdx.x * blockDim.x + threadIdx.x;
    if (idx < total) {
        int col = idx & (DIM - 1);
        float v = accum[idx] + b[col];
        v = fmaxf(v, 0.0f);                               // ReLU
        bool keep = (hash_u32((uint32_t)idx) & 1u) != 0;  // p = 0.5
        v = keep ? v * 2.0f : 0.0f;                       // / (1 - p)
        out[idx] = v + x[idx];                            // residual
    }
}

// ---------------------------------------------------------------------------
extern "C" void kernel_launch(void* const* d_in, const int* in_sizes, int n_in,
                              void* d_out, int out_size, void* d_ws, size_t ws_size,
                              hipStream_t stream) {
    const float*     x  = (const float*)d_in[0];
    const long long* ei = (const long long*)d_in[1];   // int64 edge_index [2, E]
    const float*     W  = (const float*)d_in[2];
    const float*     b  = (const float*)d_in[3];
    float*           out = (float*)d_out;

    const int N = in_sizes[0] / DIM;     // 50000
    const int E = in_sizes[1] / 2;       // 625000
    const int total = N * DIM;

    // Workspace layout (256B aligned): deg[N] | h[N*DIM] | accum[N*DIM]
    char* ws = (char*)d_ws;
    size_t off = 0;
    float* deg   = (float*)(ws + off); off += ((size_t)N * sizeof(float) + 255) & ~(size_t)255;
    float* h     = (float*)(ws + off); off += ((size_t)total * sizeof(float) + 255) & ~(size_t)255;
    float* accum = (float*)(ws + off);

    // 1) degrees (self-loop = 1, then count edge destinations)
    k_deg_init<<<(N + 255) / 256, 256, 0, stream>>>(deg, N);
    k_deg_count<<<(E + 255) / 256, 256, 0, stream>>>(ei, deg, E);

    // 2) h = x @ W  (WMMA bf16 -> f32), 64 rows per block
    int gemmBlocks = (N + ROWS_PER_BLOCK - 1) / ROWS_PER_BLOCK;
    k_gemm_wmma<<<gemmBlocks, 256, 0, stream>>>(x, W, h, N);

    // 3) self-loop seed: accum = h / deg
    k_selfloop<<<(total + 255) / 256, 256, 0, stream>>>(h, deg, accum, total);

    // 4) edge scatter-add (one wave per edge)
    long long sc_threads = (long long)E * 32;
    k_scatter<<<(int)((sc_threads + 255) / 256), 256, 0, stream>>>(ei, h, deg, accum, E);

    // 5) bias + ReLU + dropout + residual
    k_finalize<<<(total + 255) / 256, 256, 0, stream>>>(accum, b, x, out, total);
}
